// TacotronDecoder_13511967113386
// MI455X (gfx1250) — matrix operand order repaired and checked by reading, not compiled, not run on hardware
//
#include <hip/hip_runtime.h>
#include <hip/hip_bf16.h>
#include <math.h>

// ---- problem constants (from reference) ----
#define E_    512
#define D_    1024
#define A_    128
#define M_    80
#define B_    32
#define S_    256
#define T_    500
#define PRE1_ 256
#define PRE2_ 128

typedef __bf16 bf16_t;
typedef __bf16 v16bf __attribute__((ext_vector_type(16)));
typedef __bf16 v8bf  __attribute__((ext_vector_type(8)));
typedef float  v8f   __attribute__((ext_vector_type(8)));

__device__ __forceinline__ float sigm(float x) { return 1.f / (1.f + expf(-x)); }

// ============================================================================
// Weight packer: W[N,K] fp32 (row-major, "out = x @ W.T") -> bf16 B-matrix
// blocks for V_WMMA_F32_16X16X32_BF16.
// Block (ntile, ktile): 32 lanes x 16 contiguous bf16.
//   lane l (l<16):  n = ntile*16 + l,      k = ktile*32 + (e<8 ? e : 16+e-8)       (e=0..15)
//   lane l (l>=16): n = ntile*16 + l-16,   k = ktile*32 + (e<8 ? 8+e : 24+e-8)
// Multiple W's can be packed into one buffer at different ktile0 (K-concat).
// ============================================================================
__global__ void pack_w_kernel(bf16_t* __restrict__ dst, const float* __restrict__ W,
                              int Ntiles, int Kt, int KtotTiles, int ktile0,
                              int Nsrc, int Ksrc) {
  int idx = blockIdx.x * 256 + threadIdx.x;
  int total = Ntiles * Kt * 512;
  if (idx >= total) return;
  int e     = idx & 15;
  int lane  = (idx >> 4) & 31;
  int ktile = (idx >> 9) % Kt;
  int ntile = idx / (Kt * 512);
  int n = ntile * 16 + (lane & 15);
  int k = ktile * 32 + (lane >> 4) * 8 + (e < 8 ? e : 16 + (e - 8));
  float v = 0.f;
  if (n < Nsrc && k < Ksrc) v = W[(long)n * Ksrc + k];
  long o = (((long)ntile * KtotTiles + (ktile0 + ktile)) * 32 + lane) * 16 + e;
  dst[o] = (bf16_t)v;
}

// ============================================================================
// Generic bf16 WMMA GEMM (setup / final projections): one wave per 32x32 tile.
// ============================================================================
__device__ __forceinline__ void store_tile(v8f acc, int m0, int n0, int lane,
                                           const float* bias,
                                           float* Cf, int ldcf,
                                           bf16_t* Cb, int ldcb,
                                           int Nmax, int relu) {
  int lm = lane & 15, kh = lane >> 4;
  int n = n0 + lm;
  if (n >= Nmax) return;
  float bn = bias ? bias[n] : 0.f;
#pragma unroll
  for (int v = 0; v < 8; ++v) {
    int m = m0 + kh * 8 + v;
    float val = acc[v] + bn;
    if (relu) val = fmaxf(val, 0.f);
    if (Cf) Cf[(long)m * ldcf + n] = val;
    if (Cb) Cb[(long)m * ldcb + n] = (bf16_t)val;
  }
}

__global__ void __launch_bounds__(32)
gemm_bf16_wmma(const bf16_t* __restrict__ A, int lda,
               const bf16_t* __restrict__ Bp, int ktTiles,
               const float* __restrict__ bias,
               float* Cf, int ldcf, bf16_t* Cb, int ldcb,
               int Nmax, int relu) {
  const int lane = threadIdx.x;
  const int n0 = blockIdx.x * 32;
  const int m0 = blockIdx.y * 32;
  const int lm = lane & 15;
  const int kh = lane >> 4;

  v8f acc00 = {}, acc01 = {}, acc10 = {}, acc11 = {};

  const bf16_t* a0 = A + (long)(m0 + lm) * lda + kh * 8;
  const bf16_t* a1 = A + (long)(m0 + 16 + lm) * lda + kh * 8;
  const bf16_t* b0 = Bp + (((long)(n0 >> 4)    ) * ktTiles * 32 + lane) * 16;
  const bf16_t* b1 = Bp + (((long)(n0 >> 4) + 1) * ktTiles * 32 + lane) * 16;

  for (int kt = 0; kt < ktTiles; ++kt) {
    union { v16bf v; v8bf h[2]; } fa0, fa1, fb0, fb1;
    fa0.h[0] = *(const v8bf*)(a0);      fa0.h[1] = *(const v8bf*)(a0 + 16);
    fa1.h[0] = *(const v8bf*)(a1);      fa1.h[1] = *(const v8bf*)(a1 + 16);
    fb0.v = *(const v16bf*)(b0);
    fb1.v = *(const v16bf*)(b1);
    __builtin_prefetch(b0 + 512, 0, 1);
    acc00 = __builtin_amdgcn_wmma_f32_16x16x32_bf16(false, fa0.v, false, fb0.v, (short)0, acc00, false, false);
    acc01 = __builtin_amdgcn_wmma_f32_16x16x32_bf16(false, fa0.v, false, fb1.v, (short)0, acc01, false, false);
    acc10 = __builtin_amdgcn_wmma_f32_16x16x32_bf16(false, fa1.v, false, fb0.v, (short)0, acc10, false, false);
    acc11 = __builtin_amdgcn_wmma_f32_16x16x32_bf16(false, fa1.v, false, fb1.v, (short)0, acc11, false, false);
    a0 += 32; a1 += 32;
    b0 += 512; b1 += 512;
  }
  store_tile(acc00, m0,      n0,      lane, bias, Cf, ldcf, Cb, ldcb, Nmax, relu);
  store_tile(acc01, m0,      n0 + 16, lane, bias, Cf, ldcf, Cb, ldcb, Nmax, relu);
  store_tile(acc10, m0 + 16, n0,      lane, bias, Cf, ldcf, Cb, ldcb, Nmax, relu);
  store_tile(acc11, m0 + 16, n0 + 16, lane, bias, Cf, ldcf, Cb, ldcb, Nmax, relu);
}

// ============================================================================
// Fused gate-GEMM + LSTM cell, one wave per 16 hidden units (64 waves total).
// Wave w computes, for ALL 32 batch rows, the four gate columns
// { h0+j, 1024+h0+j, 2048+h0+j, 3072+h0+j } (j=0..15, h0=w*16):
// 8 accumulators = [gate][mtile]. After the K-loop, i/f/g/o for a given
// (batch, hidden) land in the SAME lane & acc element -> the LSTM cell is
// pure in-register VALU (no LDS, no second kernel, no gates round-trip).
// Callers ping-pong the h destination buffers so blocks never race readers.
// ============================================================================
__device__ __forceinline__ void
lstm_wave_body(int wave, int lane,
               const bf16_t* __restrict__ A, int lda,
               const bf16_t* __restrict__ Bp, int ktTiles,
               const float* __restrict__ bias,
               float* __restrict__ cstate,              // [32,1024]
               bf16_t* __restrict__ h1, long ld1,
               bf16_t* __restrict__ h2, long ld2) {
  const int lm = lane & 15;
  const int kh = lane >> 4;
  const int h0 = wave * 16;

  v8f acc[4][2] = {};

  const bf16_t* a0 = A + (long)lm * lda + kh * 8;
  const bf16_t* a1 = A + (long)(16 + lm) * lda + kh * 8;
  const bf16_t* bg0 = Bp + (((long)(0 * 64 + (h0 >> 4))) * ktTiles * 32 + lane) * 16;
  const bf16_t* bg1 = Bp + (((long)(1 * 64 + (h0 >> 4))) * ktTiles * 32 + lane) * 16;
  const bf16_t* bg2 = Bp + (((long)(2 * 64 + (h0 >> 4))) * ktTiles * 32 + lane) * 16;
  const bf16_t* bg3 = Bp + (((long)(3 * 64 + (h0 >> 4))) * ktTiles * 32 + lane) * 16;

  for (int kt = 0; kt < ktTiles; ++kt) {
    union { v16bf v; v8bf h[2]; } fa0, fa1;
    fa0.h[0] = *(const v8bf*)(a0);  fa0.h[1] = *(const v8bf*)(a0 + 16);
    fa1.h[0] = *(const v8bf*)(a1);  fa1.h[1] = *(const v8bf*)(a1 + 16);
    v16bf fb0 = *(const v16bf*)(bg0);
    v16bf fb1 = *(const v16bf*)(bg1);
    v16bf fb2 = *(const v16bf*)(bg2);
    v16bf fb3 = *(const v16bf*)(bg3);
    __builtin_prefetch(bg0 + 512, 0, 1);
    acc[0][0] = __builtin_amdgcn_wmma_f32_16x16x32_bf16(false, fa0.v, false, fb0, (short)0, acc[0][0], false, false);
    acc[0][1] = __builtin_amdgcn_wmma_f32_16x16x32_bf16(false, fa1.v, false, fb0, (short)0, acc[0][1], false, false);
    acc[1][0] = __builtin_amdgcn_wmma_f32_16x16x32_bf16(false, fa0.v, false, fb1, (short)0, acc[1][0], false, false);
    acc[1][1] = __builtin_amdgcn_wmma_f32_16x16x32_bf16(false, fa1.v, false, fb1, (short)0, acc[1][1], false, false);
    acc[2][0] = __builtin_amdgcn_wmma_f32_16x16x32_bf16(false, fa0.v, false, fb2, (short)0, acc[2][0], false, false);
    acc[2][1] = __builtin_amdgcn_wmma_f32_16x16x32_bf16(false, fa1.v, false, fb2, (short)0, acc[2][1], false, false);
    acc[3][0] = __builtin_amdgcn_wmma_f32_16x16x32_bf16(false, fa0.v, false, fb3, (short)0, acc[3][0], false, false);
    acc[3][1] = __builtin_amdgcn_wmma_f32_16x16x32_bf16(false, fa1.v, false, fb3, (short)0, acc[3][1], false, false);
    a0 += 32; a1 += 32;
    bg0 += 512; bg1 += 512; bg2 += 512; bg3 += 512;
  }

  // In-register LSTM cell: lane holds hidden unit hh = h0+lm for batch rows
  // m = mtile*16 + kh*8 + v  (v = acc element index).
  const int hh = h0 + lm;
  const float bi  = bias[hh];
  const float bff = bias[D_ + hh];
  const float bgg = bias[2 * D_ + hh];
  const float bo  = bias[3 * D_ + hh];
#pragma unroll
  for (int mi = 0; mi < 2; ++mi) {
#pragma unroll
    for (int v = 0; v < 8; ++v) {
      int m = mi * 16 + kh * 8 + v;  // batch row 0..31
      float ig = sigm(acc[0][mi][v] + bi);
      float fg = sigm(acc[1][mi][v] + bff);
      float gg = tanhf(acc[2][mi][v] + bgg);
      float og = sigm(acc[3][mi][v] + bo);
      long ci = (long)m * D_ + hh;
      float c = fg * cstate[ci] + ig * gg;
      cstate[ci] = c;
      bf16_t hn = (bf16_t)(og * tanhf(c));
      h1[(long)m * ld1 + hh] = hn;
      h2[(long)m * ld2 + hh] = hn;
    }
  }
}

// ============================================================================
// Attention step body (one 256-thread block per batch element):
//   dp   = ah_{t-1} @ dec_proj_w.T + b        (bf16 weights, LDS-staged ah)
//   s[s] = v . tanh(enc_proj[b,s,:] + dp) + v_b ; softmax ; context
// Stages prenet(t) and context into xbuf (the [pre|ctx|ah] GEMM A row), and
// context into the [dh|ctx] rows for the batched final projection.
// ============================================================================
__device__ __forceinline__ void
attention_body(int b, int tid, int t,
               const float* __restrict__ encp,   // [B*S, A] enc_proj fp32
               const bf16_t* __restrict__ encb,  // [B*S, E] encoder bf16
               const bf16_t* __restrict__ wdp,   // [A, D] dec_proj_w bf16
               const float* __restrict__ dpb,
               const float* __restrict__ vw, const float* __restrict__ vb,
               const bf16_t* __restrict__ pre,   // [B*T, PRE2]
               bf16_t* __restrict__ xbuf,        // [B,1664]=[pre|ctx|ah_{t-1}]
               bf16_t* __restrict__ dob,         // [B*T, 1536] = [dh|ctx]
               float* __restrict__ aw_out) {
  __shared__ float ah_s[D_];
  __shared__ float dps[A_], vs[A_], red[S_], aws[S_];

  // ah_{t-1} row -> LDS (read-only region of xbuf; writes below are disjoint)
  for (int k = tid; k < D_; k += 256) ah_s[k] = (float)xbuf[(long)b * 1664 + 640 + k];
  __syncthreads();

  // dp: 2 threads per output a, each sums half of K=1024
  {
    int a = tid & 127, half = tid >> 7;
    const bf16_t* wr = wdp + (long)a * D_ + half * 512;
    const float* ar = ah_s + half * 512;
    float s = 0.f;
#pragma unroll 4
    for (int k = 0; k < 512; ++k) s += ar[k] * (float)wr[k];
    red[tid] = s;
  }
  __syncthreads();
  if (tid < A_) { dps[tid] = red[tid] + red[tid + 128] + dpb[tid]; vs[tid] = vw[tid]; }
  __syncthreads();

  // scores
  const float* ep = encp + ((long)b * S_ + tid) * A_;
  float acc = 0.f;
#pragma unroll 4
  for (int a = 0; a < A_; ++a) acc += tanhf(ep[a] + dps[a]) * vs[a];
  float sc = acc + vb[0];

  red[tid] = sc; __syncthreads();
  for (int off = 128; off; off >>= 1) { if (tid < off) red[tid] = fmaxf(red[tid], red[tid + off]); __syncthreads(); }
  float mx = red[0]; __syncthreads();
  float ex = expf(sc - mx);
  red[tid] = ex; __syncthreads();
  for (int off = 128; off; off >>= 1) { if (tid < off) red[tid] += red[tid + off]; __syncthreads(); }
  float aw = ex / red[0];
  aws[tid] = aw;
  aw_out[((long)b * T_ + t) * S_ + tid] = aw;
  if (tid < PRE2_) xbuf[(long)b * 1664 + tid] = pre[((long)b * T_ + t) * PRE2_ + tid];
  __syncthreads();

  // context[e] = sum_s aw[s] * enc[b,s,e]  (lanes stride e -> coalesced)
  for (int e = tid; e < E_; e += 256) {
    float c = 0.f;
    const bf16_t* eb = encb + (long)b * S_ * E_ + e;
    for (int s = 0; s < S_; ++s) c += aws[s] * (float)eb[(long)s * E_];
    bf16_t cb = (bf16_t)c;
    xbuf[(long)b * 1664 + PRE2_ + e] = cb;
    dob[((long)b * T_ + t) * 1536 + D_ + e] = cb;
  }
}

// ---- kernel wrappers ----

// standalone attention (primes t=0)
__global__ void __launch_bounds__(256)
attn_kernel(const float* encp, const bf16_t* encb, const bf16_t* wdp,
            const float* dpb, const float* vw, const float* vb,
            const bf16_t* pre, bf16_t* xbuf, bf16_t* dob, float* aw_out, int t) {
  attention_body(blockIdx.x, threadIdx.x, t, encp, encb, wdp, dpb, vw, vb,
                 pre, xbuf, dob, aw_out);
}

// attention-LSTM: 8 blocks x 8 waves = 64 waves (K=1664 gate GEMM + cell)
__global__ void __launch_bounds__(256)
attn_lstm_kernel(const bf16_t* __restrict__ A, const bf16_t* __restrict__ Bp,
                 const float* __restrict__ bias, float* __restrict__ cstate,
                 bf16_t* __restrict__ h1, bf16_t* __restrict__ h2) {
  int wave = blockIdx.x * 8 + (threadIdx.x >> 5);
  lstm_wave_body(wave, threadIdx.x & 31, A, 1664, Bp, 52, bias, cstate,
                 h1, 2048, h2, 1664);
}

// Heterogeneous kernel: blocks 0..7 run the decoder-LSTM for step t (8 waves
// each); blocks 8..39 run attention for step t+1 (independent: it needs only
// ah_t, produced by the attention-LSTM kernel that precedes this one).
__global__ void __launch_bounds__(256)
declstm_attn_kernel(const bf16_t* __restrict__ Ay,   // Y[t%2] = [ah_t|dh_{t-1}]
                    const bf16_t* __restrict__ BpDec,
                    const float* __restrict__ decb, float* __restrict__ dc,
                    bf16_t* __restrict__ yh1,        // Y[(t+1)%2] dh slot
                    bf16_t* __restrict__ doh, long dold, // DOB row t, dh slot
                    const float* __restrict__ encp, const bf16_t* __restrict__ encb,
                    const bf16_t* __restrict__ wdp, const float* __restrict__ dpb,
                    const float* __restrict__ vw, const float* __restrict__ vb,
                    const bf16_t* __restrict__ pre,
                    bf16_t* __restrict__ xnext,      // X[(t+1)%2]
                    bf16_t* __restrict__ dob, float* __restrict__ aw_out,
                    int tp1, int do_attn) {
  if (blockIdx.x < 8) {
    int wave = blockIdx.x * 8 + (threadIdx.x >> 5);
    lstm_wave_body(wave, threadIdx.x & 31, Ay, 2048, BpDec, 64, decb, dc,
                   yh1, 2048, doh, dold);
  } else if (do_attn) {
    attention_body(blockIdx.x - 8, threadIdx.x, tp1, encp, encb, wdp, dpb,
                   vw, vb, pre, xnext, dob, aw_out);
  }
}

// ============================================================================
// Small elementwise helpers
// ============================================================================
__global__ void f32_to_bf16_kernel(bf16_t* __restrict__ dst, const float* __restrict__ src, int n) {
  int i = blockIdx.x * 256 + threadIdx.x;
  if (i < n) dst[i] = (bf16_t)src[i];
}

__global__ void add_bias_kernel(float* __restrict__ dst, const float* __restrict__ a,
                                const float* __restrict__ b, int n) {
  int i = blockIdx.x * 256 + threadIdx.x;
  if (i < n) dst[i] = a[i] + b[i];
}

// Teacher-forced prenet input: rows r=b*T+t of [16000 x 96] (K padded 80->96)
__global__ void pack_a1_kernel(bf16_t* __restrict__ dst, const float* __restrict__ targets) {
  int idx = blockIdx.x * 256 + threadIdx.x;
  if (idx >= 16000 * 96) return;
  int r = idx / 96, k = idx - r * 96;
  int b = r / T_, t = r - b * T_;
  float v = 0.f;
  if (k < M_ && t > 0) v = targets[((long)b * T_ + (t - 1)) * M_ + k];
  dst[idx] = (bf16_t)v;
}

// stop token: sigmoid([dh|ctx] . stop_w + b), one block per (b,t) row
__global__ void __launch_bounds__(128)
stop_kernel(const bf16_t* __restrict__ dob, const float* __restrict__ sw,
            const float* __restrict__ sb, float* __restrict__ out) {
  int r = blockIdx.x, tid = threadIdx.x;
  __shared__ float red[128];
  const bf16_t* row = dob + (long)r * 1536;
  float acc = 0.f;
  for (int k = tid; k < 1536; k += 128) acc += (float)row[k] * sw[k];
  red[tid] = acc; __syncthreads();
  for (int off = 64; off; off >>= 1) { if (tid < off) red[tid] += red[tid + off]; __syncthreads(); }
  if (tid == 0) out[r] = sigm(red[0] + sb[0]);
}

// ============================================================================
// Host orchestration
// ============================================================================
extern "C" void kernel_launch(void* const* d_in, const int* in_sizes, int n_in,
                              void* d_out, int out_size, void* d_ws, size_t ws_size,
                              hipStream_t stream) {
  const float* enc        = (const float*)d_in[0];   // [32,256,512]
  const float* targets    = (const float*)d_in[1];   // [32,500,80]
  const float* enc_proj_w = (const float*)d_in[2];   // [128,512]
  const float* enc_proj_b = (const float*)d_in[3];
  const float* dec_proj_w = (const float*)d_in[4];   // [128,1024]
  const float* dec_proj_b = (const float*)d_in[5];
  const float* v_w        = (const float*)d_in[6];   // [1,128]
  const float* v_b        = (const float*)d_in[7];
  const float* pre_w1     = (const float*)d_in[8];   // [256,80]
  const float* pre_b1     = (const float*)d_in[9];
  const float* pre_w2     = (const float*)d_in[10];  // [128,256]
  const float* pre_b2     = (const float*)d_in[11];
  const float* attn_ih_w  = (const float*)d_in[12];  // [4096,640]
  const float* attn_ih_b  = (const float*)d_in[13];
  const float* attn_hh_w  = (const float*)d_in[14];  // [4096,1024]
  const float* attn_hh_b  = (const float*)d_in[15];
  const float* dec_ih_w   = (const float*)d_in[16];  // [4096,1024]
  const float* dec_ih_b   = (const float*)d_in[17];
  const float* dec_hh_w   = (const float*)d_in[18];  // [4096,1024]
  const float* dec_hh_b   = (const float*)d_in[19];
  const float* proj_w     = (const float*)d_in[20];  // [80,1536]
  const float* proj_b     = (const float*)d_in[21];
  const float* stop_w     = (const float*)d_in[22];  // [1,1536]
  const float* stop_b     = (const float*)d_in[23];

  float* out_mel  = (float*)d_out;                       // [32,500,80]
  float* out_attn = out_mel + (long)B_ * T_ * M_;        // [32,500,256]
  float* out_stop = out_attn + (long)B_ * T_ * S_;       // [32,500,1]

  // ---- workspace carve-up (bf16 weights ~30MB stay hot in 192MB L2) ----
  char* p = (char*)d_ws;
  auto take = [&](size_t bytes) { char* r = p; p += (bytes + 255) & ~(size_t)255; return r; };
  bf16_t* PK_PRE1 = (bf16_t*)take((size_t)256 * 96 * 2);
  bf16_t* PK_PRE2 = (bf16_t*)take((size_t)128 * 256 * 2);
  bf16_t* PK_ENC  = (bf16_t*)take((size_t)128 * 512 * 2);
  bf16_t* PK_ATTN = (bf16_t*)take((size_t)4096 * 1664 * 2);
  bf16_t* PK_DEC  = (bf16_t*)take((size_t)4096 * 2048 * 2);
  bf16_t* PK_PROJ = (bf16_t*)take((size_t)96 * 1536 * 2);
  bf16_t* WDPB    = (bf16_t*)take((size_t)128 * 1024 * 2);   // dec_proj_w bf16
  bf16_t* A1      = (bf16_t*)take((size_t)16000 * 96 * 2);
  bf16_t* H1      = (bf16_t*)take((size_t)16000 * 256 * 2);
  bf16_t* PRE     = (bf16_t*)take((size_t)16000 * 128 * 2);
  bf16_t* ENCB    = (bf16_t*)take((size_t)8192 * 512 * 2);
  float*  ENCP    = (float*)take((size_t)8192 * 128 * 4);
  bf16_t* XBUF[2]; // ping-pong [pre|ctx|ah] (race-free h writes in fused LSTM)
  XBUF[0] = (bf16_t*)take((size_t)32 * 1664 * 2);
  XBUF[1] = (bf16_t*)take((size_t)32 * 1664 * 2);
  bf16_t* YBUF[2]; // ping-pong [ah|dh]
  YBUF[0] = (bf16_t*)take((size_t)32 * 2048 * 2);
  YBUF[1] = (bf16_t*)take((size_t)32 * 2048 * 2);
  float*  AC      = (float*)take((size_t)32 * 1024 * 4);
  float*  DC      = (float*)take((size_t)32 * 1024 * 4);
  float*  ATTNB   = (float*)take((size_t)4096 * 4);
  float*  DECB    = (float*)take((size_t)4096 * 4);
  bf16_t* DOB     = (bf16_t*)take((size_t)16000 * 1536 * 2); // [dh|ctx]

  // ---- zero recurrent state (deterministic per call) ----
  hipMemsetAsync(XBUF[0], 0, (size_t)32 * 1664 * 2, stream);
  hipMemsetAsync(XBUF[1], 0, (size_t)32 * 1664 * 2, stream);
  hipMemsetAsync(YBUF[0], 0, (size_t)32 * 2048 * 2, stream);
  hipMemsetAsync(YBUF[1], 0, (size_t)32 * 2048 * 2, stream);
  hipMemsetAsync(AC, 0, (size_t)32 * 1024 * 4, stream);
  hipMemsetAsync(DC, 0, (size_t)32 * 1024 * 4, stream);

  auto grid1d = [](long n) { return dim3((unsigned)((n + 255) / 256)); };

  // ---- one-time (per launch) weight packing ----
  pack_w_kernel<<<grid1d((long)16 * 3 * 512), 256, 0, stream>>>(PK_PRE1, pre_w1, 16, 3, 3, 0, 256, 80);
  pack_w_kernel<<<grid1d((long)8  * 8 * 512), 256, 0, stream>>>(PK_PRE2, pre_w2, 8, 8, 8, 0, 128, 256);
  pack_w_kernel<<<grid1d((long)8  * 16 * 512), 256, 0, stream>>>(PK_ENC, enc_proj_w, 8, 16, 16, 0, 128, 512);
  pack_w_kernel<<<grid1d((long)256 * 20 * 512), 256, 0, stream>>>(PK_ATTN, attn_ih_w, 256, 20, 52, 0, 4096, 640);
  pack_w_kernel<<<grid1d((long)256 * 32 * 512), 256, 0, stream>>>(PK_ATTN, attn_hh_w, 256, 32, 52, 20, 4096, 1024);
  pack_w_kernel<<<grid1d((long)256 * 32 * 512), 256, 0, stream>>>(PK_DEC, dec_ih_w, 256, 32, 64, 0, 4096, 1024);
  pack_w_kernel<<<grid1d((long)256 * 32 * 512), 256, 0, stream>>>(PK_DEC, dec_hh_w, 256, 32, 64, 32, 4096, 1024);
  pack_w_kernel<<<grid1d((long)6 * 48 * 512), 256, 0, stream>>>(PK_PROJ, proj_w, 6, 48, 48, 0, 80, 1536);
  f32_to_bf16_kernel<<<grid1d((long)128 * 1024), 256, 0, stream>>>(WDPB, dec_proj_w, 128 * 1024);
  add_bias_kernel<<<16, 256, 0, stream>>>(ATTNB, attn_ih_b, attn_hh_b, 4096);
  add_bias_kernel<<<16, 256, 0, stream>>>(DECB, dec_ih_b, dec_hh_b, 4096);

  // ---- hoisted out of the recurrence ----
  f32_to_bf16_kernel<<<grid1d((long)8192 * 512), 256, 0, stream>>>(ENCB, enc, 8192 * 512);
  gemm_bf16_wmma<<<dim3(4, 256), 32, 0, stream>>>(ENCB, 512, PK_ENC, 16, enc_proj_b,
                                                  ENCP, 128, (bf16_t*)nullptr, 0, 128, 0);
  pack_a1_kernel<<<grid1d((long)16000 * 96), 256, 0, stream>>>(A1, targets);
  gemm_bf16_wmma<<<dim3(8, 500), 32, 0, stream>>>(A1, 96, PK_PRE1, 3, pre_b1,
                                                  (float*)nullptr, 0, H1, 256, 256, 1);
  gemm_bf16_wmma<<<dim3(4, 500), 32, 0, stream>>>(H1, 256, PK_PRE2, 8, pre_b2,
                                                  (float*)nullptr, 0, PRE, 128, 128, 1);

  // ---- sequential recurrence: 2 fused kernels per step ----
  // prime attention for t=0 (ah_{-1}=0 already in XBUF[0])
  attn_kernel<<<B_, 256, 0, stream>>>(ENCP, ENCB, WDPB, dec_proj_b,
                                      v_w, v_b, PRE, XBUF[0], DOB, out_attn, 0);
  for (int t = 0; t < T_; ++t) {
    bf16_t* Xc = XBUF[t & 1];       // [pre_t|ctx_t|ah_{t-1}]
    bf16_t* Xn = XBUF[(t + 1) & 1]; // receives ah_t then pre/ctx_{t+1}
    bf16_t* Yc = YBUF[t & 1];       // receives ah_t; holds dh_{t-1}
    bf16_t* Yn = YBUF[(t + 1) & 1]; // receives dh_t

    // attention-LSTM: gates GEMM (K=1664) + cell; ah_t -> Yc & Xn.ah
    attn_lstm_kernel<<<8, 256, 0, stream>>>(Xc, PK_ATTN, ATTNB, AC, Yc, Xn + 640);
    // decoder-LSTM for step t  ||  attention for step t+1 (uses ah_t from Xn)
    declstm_attn_kernel<<<40, 256, 0, stream>>>(Yc, PK_DEC, DECB, DC,
                                                Yn + 1024,
                                                DOB + (long)t * 1536, (long)T_ * 1536,
                                                ENCP, ENCB, WDPB, dec_proj_b,
                                                v_w, v_b, PRE, Xn, DOB, out_attn,
                                                t + 1, (t + 1 < T_) ? 1 : 0);
  }

  // ---- output projections, batched over all 500 steps (M=16000,K=1536) ----
  gemm_bf16_wmma<<<dim3(3, 500), 32, 0, stream>>>(DOB, 1536, PK_PROJ, 48, proj_b,
                                                  out_mel, 80, (bf16_t*)nullptr, 0, 80, 0);
  stop_kernel<<<16000, 128, 0, stream>>>(DOB, stop_w, stop_b, out_stop);
}
